// TransformerBlock_77360950935618
// MI455X (gfx1250) — matrix operand order repaired and checked by reading, not compiled, not run on hardware
//
#include <hip/hip_runtime.h>
#include <hip/hip_bf16.h>
#include <math.h>

typedef __attribute__((ext_vector_type(16))) _Float16 v16h;
typedef __attribute__((ext_vector_type(8)))  _Float16 v8h;
typedef __attribute__((ext_vector_type(8)))  float    v8f;

// global-address-space pointer types: force GLOBAL_LOAD (not FLAT_LOAD) so
// vector loads don't tick DScnt / the LDS-aperture path (CDNA5 FLAT uses both
// LOADcnt and DScnt; keep DS exclusively for the attention P-transpose).
typedef const __attribute__((address_space(1))) v8h*  gv8h_p;
typedef const __attribute__((address_space(1))) v16h* gv16h_p;

#define DIMC   320
#define HEADS  8
#define DH     40
#define DHP    64
#define SEQL   4096
#define NB     2
#define KVD    768
#define KVROWS 80     // 77 padded to 80 (mult of 16)
#define LKP2   96     // cross-attn key pad (mult of 32)

// ---------------------------------------------------------------------------
// helpers
// ---------------------------------------------------------------------------
__device__ __forceinline__ v16h cat8(v8h lo, v8h hi) {
  return __builtin_shufflevector(lo, hi, 0,1,2,3,4,5,6,7,8,9,10,11,12,13,14,15);
}
// contiguous 32B global load (B fragment: lane-contiguous packed layout)
__device__ __forceinline__ v16h gload16(const _Float16* p) {
  return *(gv16h_p)p;
}
// A-fragment (16x32 f16) loader from row-major global memory.
// caller passes p = base + row*ld + (half?16:0); lanes 0-15: K 0..15,
// lanes 16-31: K 16..31 (contiguous, matching B fragments)
__device__ __forceinline__ v16h load_afrag(const _Float16* p) {
  v8h lo = *(gv8h_p)p;
  v8h hi = *(gv8h_p)(p + 8);
  return cat8(lo, hi);
}
__device__ __forceinline__ v8f wmma32(v16h a, v16h b, v8f c) {
  return __builtin_amdgcn_wmma_f32_16x16x32_f16(false, a, false, b, (short)0, c,
                                                false, false);
}

// ---------------------------------------------------------------------------
// weight packing: f32 [K,N] row-major -> f16 B-fragment tiles (32K x 16N).
// B layout (contiguous k order, same as load_afrag's A): lane l holds column
// n = nt*16 + (l&15); element e (0..15) holds K = kt*32 + (l>>4)*16 + e.
// Tile (kt,nt) stored at (nt*ktiles + kt)*512, lane-major, 16 f16 per lane.
// ---------------------------------------------------------------------------
__global__ __launch_bounds__(256)
void pack_b_kernel(const float* __restrict__ W, _Float16* __restrict__ Bp,
                   int K, int N) {
  int idx = blockIdx.x * 256 + threadIdx.x;
  if (idx >= K * N) return;
  int e    = idx & 15;
  int lane = (idx >> 4) & 31;
  int tile = idx >> 9;
  int ktiles = K >> 5;
  int nt = tile / ktiles;
  int kt = tile - nt * ktiles;
  int n = nt * 16 + (lane & 15);
  int k = kt * 32 + (lane >> 4) * 16 + e;
  Bp[idx] = (_Float16)W[(size_t)k * N + n];
}

// ---------------------------------------------------------------------------
// GroupNorm statistics: one block per (b,group); group = 10 channels * 4096
// contiguous floats.
// ---------------------------------------------------------------------------
__global__ __launch_bounds__(256)
void gn_stats_kernel(const float* __restrict__ q, float* __restrict__ mu,
                     float* __restrict__ rsig, int ge, float eps) {
  __shared__ float sh[16];
  int g = blockIdx.x;
  const float* p = q + (size_t)g * ge;
  float s = 0.f, ss = 0.f;
  for (int i = threadIdx.x; i < ge; i += 256) { float v = p[i]; s += v; ss += v * v; }
  for (int o = 16; o; o >>= 1) { s += __shfl_xor(s, o); ss += __shfl_xor(ss, o); }
  int w = threadIdx.x >> 5;
  if ((threadIdx.x & 31) == 0) { sh[w] = s; sh[8 + w] = ss; }
  __syncthreads();
  if (threadIdx.x == 0) {
    s = 0.f; ss = 0.f;
    for (int i = 0; i < 8; ++i) { s += sh[i]; ss += sh[8 + i]; }
    float m = s / ge;
    float var = ss / ge - m * m;
    mu[g] = m;
    rsig[g] = rsqrtf(var + eps);
  }
}

// GroupNorm apply + transpose (b,c,hw) -> f16 (b,hw,c)
__global__ __launch_bounds__(256)
void gn_apply_kernel(const float* __restrict__ q, const float* __restrict__ mu,
                     const float* __restrict__ rsig, const float* __restrict__ gs,
                     const float* __restrict__ gb, _Float16* __restrict__ xn) {
  int bz = blockIdx.z;
  size_t idx = (size_t)blockIdx.x * 256 + threadIdx.x;   // over DIMC*SEQL
  if (idx >= (size_t)DIMC * SEQL) return;
  int c = (int)(idx / SEQL);
  int m = (int)(idx - (size_t)c * SEQL);
  int g = bz * 32 + c / 10;
  float v = q[((size_t)bz * DIMC + c) * SEQL + m];
  v = (v - mu[g]) * rsig[g] * gs[c] + gb[c];
  xn[((size_t)bz * SEQL + m) * DIMC + c] = (_Float16)v;
}

// LayerNorm: one block (128 thr) per row of 320; f32 in -> f16 out
__global__ __launch_bounds__(128)
void layernorm_kernel(const float* __restrict__ x, const float* __restrict__ sc,
                      const float* __restrict__ bi, _Float16* __restrict__ y,
                      float eps) {
  __shared__ float sh[8];
  __shared__ float stats[2];
  size_t row = blockIdx.x;
  const float* xr = x + row * DIMC;
  float s = 0.f, ss = 0.f;
  for (int i = threadIdx.x; i < DIMC; i += 128) { float v = xr[i]; s += v; ss += v * v; }
  for (int o = 16; o; o >>= 1) { s += __shfl_xor(s, o); ss += __shfl_xor(ss, o); }
  int w = threadIdx.x >> 5;
  if ((threadIdx.x & 31) == 0) { sh[w] = s; sh[4 + w] = ss; }
  __syncthreads();
  if (threadIdx.x == 0) {
    s = sh[0] + sh[1] + sh[2] + sh[3];
    ss = sh[4] + sh[5] + sh[6] + sh[7];
    float m = s / DIMC;
    float var = ss / DIMC - m * m;
    stats[0] = m;
    stats[1] = rsqrtf(var + eps);
  }
  __syncthreads();
  float m = stats[0], r = stats[1];
  _Float16* yr = y + row * DIMC;
  for (int i = threadIdx.x; i < DIMC; i += 128)
    yr[i] = (_Float16)((xr[i] - m) * r * sc[i] + bi[i]);
}

// kv f32 (b,77,768) -> f16 (b,80,768), rows 77..79 zeroed
__global__ __launch_bounds__(256)
void kv_cvt_kernel(const float* __restrict__ kv, _Float16* __restrict__ kv16) {
  size_t idx = (size_t)blockIdx.x * 256 + threadIdx.x;
  if (idx >= (size_t)NB * KVROWS * KVD) return;
  int bz  = (int)(idx / (KVROWS * KVD));
  int rem = (int)(idx - (size_t)bz * KVROWS * KVD);
  int row = rem / KVD, col = rem - row * KVD;
  float v = (row < 77) ? kv[((size_t)bz * 77 + row) * KVD + col] : 0.f;
  kv16[idx] = (_Float16)v;
}

// write a column of ones at head-dim d=40 of transposed-V so the P@V WMMA
// also accumulates the online-softmax denominator for free.
__global__ __launch_bounds__(256)
void vt_ones_kernel(_Float16* __restrict__ vT, int Lkp) {
  int idx = blockIdx.x * 256 + threadIdx.x;  // over NB*HEADS*Lkp
  if (idx >= NB * HEADS * Lkp) return;
  int bh = idx / Lkp, key = idx - bh * Lkp;
  vT[((size_t)bh * DHP + DH) * Lkp + key] = (_Float16)1.0f;
}

// GEGLU: h = g[:,:1280] * gelu_exact(g[:,1280:])
__global__ __launch_bounds__(256)
void geglu_kernel(const _Float16* __restrict__ g, _Float16* __restrict__ h) {
  int bz = blockIdx.z;
  size_t idx = (size_t)blockIdx.x * 256 + threadIdx.x;   // over SEQL*1280
  if (idx >= (size_t)SEQL * 1280) return;
  size_t m = idx / 1280;
  int i = (int)(idx - m * 1280);
  const _Float16* gr = g + ((size_t)bz * SEQL + m) * 2560;
  float a = (float)gr[i];
  float x = (float)gr[1280 + i];
  float ge = 0.5f * x * (1.0f + erff(x * 0.70710678118654752f));
  h[(size_t)bz * SEQL * 1280 + idx] = (_Float16)(a * ge);
}

// ---------------------------------------------------------------------------
// Generic WMMA GEMM:  C = A(MxK,f16,row-major) @ Bpacked(KxN) [+bias][+resid]
// block = 256 thr = 8 waves (4M x 2N); wave tile 64x32; block tile 256x64.
// 8 WMMAs per 12 b128 loads per K-step. N mult of 64, K mult of 32.
// ---------------------------------------------------------------------------
enum { EPI_F32 = 0, EPI_F16 = 1, EPI_QK = 2, EPI_VT = 3, EPI_TOUT = 4 };

template <int EPI>
__global__ __launch_bounds__(256)
void gemm16_kernel(const _Float16* __restrict__ A, const _Float16* __restrict__ Bp,
                   const float* __restrict__ bias, const float* __restrict__ resid,
                   void* __restrict__ Cp, int M, int N, int K, int Mstore,
                   long long sA, long long sC, long long sR, int e0) {
  int lane = threadIdx.x & 31;
  int wv = threadIdx.x >> 5;
  int wm = wv >> 1, wn = wv & 1;        // 4M x 2N waves
  int bz = blockIdx.z;
  int bm = blockIdx.y * 256 + wm * 64;
  int bn0 = blockIdx.x * 64 + wn * 32;  // two 16-col subtiles per wave
  int r = lane & 15, half = lane >> 4;
  int ktiles = K >> 5;
  const _Float16* Ab = A + (size_t)bz * sA;
  const _Float16* bp0 = Bp + ((size_t)(bn0 >> 4) * ktiles) * 512 + lane * 16;
  const _Float16* bp1 = bp0 + (size_t)ktiles * 512;
  const _Float16* arow[4];
#pragma unroll
  for (int t = 0; t < 4; ++t) {
    int row = bm + t * 16 + r;
    if (row > M - 1) row = M - 1;
    arow[t] = Ab + (size_t)row * K + half * 16;
  }
  v8f acc[4][2] = {};
  for (int kt = 0; kt < ktiles; ++kt) {
    v16h b0 = gload16(bp0); bp0 += 512;
    v16h b1 = gload16(bp1); bp1 += 512;
#pragma unroll
    for (int t = 0; t < 4; ++t) {
      v16h a = load_afrag(arow[t]);
      arow[t] += 32;
      acc[t][0] = wmma32(a, b0, acc[t][0]);
      acc[t][1] = wmma32(a, b1, acc[t][1]);
    }
  }
#pragma unroll
  for (int j = 0; j < 2; ++j) {
    int col = bn0 + j * 16 + r;
    float bb = bias ? bias[col] : 0.f;
    int head = col / DH, dd = col - head * DH;
#pragma unroll
    for (int t = 0; t < 4; ++t) {
#pragma unroll
      for (int v = 0; v < 8; ++v) {
        int row = bm + t * 16 + half * 8 + v;
        if (row >= Mstore) continue;
        float val = acc[t][j][v] + bb;
        if (EPI == EPI_F32) {
          float* C = (float*)Cp;
          if (resid) val += resid[(size_t)bz * sR + (size_t)row * N + col];
          C[(size_t)bz * sC + (size_t)row * N + col] = val;
        } else if (EPI == EPI_F16) {
          _Float16* C = (_Float16*)Cp;
          if (resid) val += resid[(size_t)bz * sR + (size_t)row * N + col];
          C[(size_t)bz * sC + (size_t)row * N + col] = (_Float16)val;
        } else if (EPI == EPI_QK) {
          _Float16* C = (_Float16*)Cp;
          C[(((size_t)(bz * HEADS + head)) * e0 + row) * DHP + dd] = (_Float16)val;
        } else if (EPI == EPI_VT) {
          _Float16* C = (_Float16*)Cp;
          C[(((size_t)(bz * HEADS + head)) * DHP + dd) * e0 + row] = (_Float16)val;
        } else {  // EPI_TOUT: out (b, N, e0) + bias + residual(q)
          float* C = (float*)Cp;
          size_t idx = ((size_t)bz * N + col) * (size_t)e0 + row;
          C[idx] = val + resid[idx];
        }
      }
    }
  }
}

// ---------------------------------------------------------------------------
// Flash attention. qp: (b,H,SEQL,64) f16 zero-padded; kp: (b,H,Lkp,64);
// vT: (b,H,64,Lkp) with ones at d=40; out: (b,SEQL,320) f16.
// 128 thr = 4 waves, 16 q-rows/wave. MASKED=false for self-attn hot path.
// ---------------------------------------------------------------------------
template <bool MASKED>
__global__ __launch_bounds__(128)
void attn_kernel(const _Float16* __restrict__ qp, const _Float16* __restrict__ kp,
                 const _Float16* __restrict__ vT, _Float16* __restrict__ out,
                 int Lk, int Lkp, float scale) {
  __shared__ _Float16 pbuf[4][16][32];
  int lane = threadIdx.x & 31, wave = threadIdx.x >> 5;
  int r = lane & 15, half = lane >> 4;
  int h = blockIdx.y, bz = blockIdx.z;
  int m0 = blockIdx.x * 64 + wave * 16;
  const _Float16* qh = qp + ((size_t)(bz * HEADS + h)) * SEQL * DHP;
  const _Float16* kh = kp + ((size_t)(bz * HEADS + h)) * Lkp * DHP;
  const _Float16* vh = vT + ((size_t)(bz * HEADS + h)) * DHP * Lkp;

  v16h aq0 = load_afrag(qh + (size_t)(m0 + r) * DHP + half * 16);
  v16h aq1 = load_afrag(qh + (size_t)(m0 + r) * DHP + 32 + half * 16);

  v8f o0 = {}, o1 = {}, o2 = {}, o3 = {};
  float Mr[8];
#pragma unroll
  for (int v = 0; v < 8; ++v) Mr[v] = -1e30f;

  int nkt = Lkp >> 5;
  for (int kt = 0; kt < nkt; ++kt) {
    int kb = kt * 32;
    // scores: two 16-key column tiles
    v8f s0 = {}, s1 = {};
    {
      const _Float16* k0 = kh + (size_t)(kb + r) * DHP + half * 16;
      s0 = wmma32(aq0, gload16(k0), s0);
      s0 = wmma32(aq1, gload16(k0 + 32), s0);
      const _Float16* k1 = kh + (size_t)(kb + 16 + r) * DHP + half * 16;
      s1 = wmma32(aq0, gload16(k1), s1);
      s1 = wmma32(aq1, gload16(k1 + 32), s1);
    }
    int c0 = kb + r, c1 = kb + 16 + r;
#pragma unroll
    for (int v = 0; v < 8; ++v) {
      float a = s0[v] * scale;
      float b = s1[v] * scale;
      if (MASKED) {
        if (c0 >= Lk) a = -1e30f;
        if (c1 >= Lk) b = -1e30f;
      }
      float t = fmaxf(a, b);
      t = fmaxf(t, __shfl_xor(t, 1));
      t = fmaxf(t, __shfl_xor(t, 2));
      t = fmaxf(t, __shfl_xor(t, 4));
      t = fmaxf(t, __shfl_xor(t, 8));
      float mn = fmaxf(Mr[v], t);
      float alpha = __expf(Mr[v] - mn);
      Mr[v] = mn;
      float p0 = __expf(a - mn), p1 = __expf(b - mn);
      o0[v] *= alpha; o1[v] *= alpha; o2[v] *= alpha; o3[v] *= alpha;
      pbuf[wave][half * 8 + v][r] = (_Float16)p0;
      pbuf[wave][half * 8 + v][16 + r] = (_Float16)p1;
    }
    // transpose P through LDS into A-fragment order (same-wave DS ordering)
    v16h pa = cat8(*(const v8h*)&pbuf[wave][r][half * 16],
                   *(const v8h*)&pbuf[wave][r][half * 16 + 8]);
    const _Float16* vb = vh + (size_t)r * Lkp + kb + half * 16;
    o0 = wmma32(pa, gload16(vb), o0);
    o1 = wmma32(pa, gload16(vb + (size_t)16 * Lkp), o1);
    o2 = wmma32(pa, gload16(vb + (size_t)32 * Lkp), o2);
    o3 = wmma32(pa, gload16(vb + (size_t)48 * Lkp), o3);
  }
  // row-sum lives in the ones column: tile nt=2, r=8, matching lane half
  int srcLane = (lane & 16) + 8;
#pragma unroll
  for (int v = 0; v < 8; ++v) {
    float inv = 1.0f / __shfl(o2[v], srcLane);
    int row = m0 + half * 8 + v;
    _Float16* orow = out + ((size_t)bz * SEQL + row) * DIMC + h * DH;
    orow[r] = (_Float16)(o0[v] * inv);
    if (16 + r < DH) orow[16 + r] = (_Float16)(o1[v] * inv);
    if (32 + r < DH) orow[32 + r] = (_Float16)(o2[v] * inv);
    // cols 40..63 are padding (40 = denominator column), never stored
  }
}

// ---------------------------------------------------------------------------
// host launcher
// ---------------------------------------------------------------------------
extern "C" void kernel_launch(void* const* d_in, const int* in_sizes, int n_in,
                              void* d_out, int out_size, void* d_ws, size_t ws_size,
                              hipStream_t stream) {
  (void)in_sizes; (void)n_in; (void)out_size; (void)ws_size;
  const float* q_in  = (const float*)d_in[0];
  const float* kv    = (const float*)d_in[1];
  const float* gn_s  = (const float*)d_in[2];
  const float* gn_b  = (const float*)d_in[3];
  const float* cin_w = (const float*)d_in[4];
  const float* cin_b = (const float*)d_in[5];
  const float* ln0_s = (const float*)d_in[6];
  const float* ln0_b = (const float*)d_in[7];
  const float* a1_wq = (const float*)d_in[8];
  const float* a1_wk = (const float*)d_in[9];
  const float* a1_wv = (const float*)d_in[10];
  const float* a1_wo = (const float*)d_in[11];
  const float* a1_bo = (const float*)d_in[12];
  const float* ln1_s = (const float*)d_in[13];
  const float* ln1_b = (const float*)d_in[14];
  const float* a2_wq = (const float*)d_in[15];
  const float* a2_wk = (const float*)d_in[16];
  const float* a2_wv = (const float*)d_in[17];
  const float* a2_wo = (const float*)d_in[18];
  const float* a2_bo = (const float*)d_in[19];
  const float* lnA_s = (const float*)d_in[20];
  const float* lnA_b = (const float*)d_in[21];
  const float* fc0_w = (const float*)d_in[22];
  const float* fc0_b = (const float*)d_in[23];
  const float* fc1_w = (const float*)d_in[24];
  const float* fc1_b = (const float*)d_in[25];
  const float* cout_w = (const float*)d_in[26];
  const float* cout_b = (const float*)d_in[27];

  char* ws = (char*)d_ws;
  size_t off = 0;
  auto alloc = [&](size_t bytes) -> void* {
    void* p = (void*)(ws + off);
    off += (bytes + 255) & ~(size_t)255;
    return p;
  };

  const size_t WB2 = (size_t)DIMC * DIMC * 2;          // 320x320 f16
  _Float16* Wcin = (_Float16*)alloc(WB2);
  _Float16* Wa1q = (_Float16*)alloc(WB2);
  _Float16* Wa1k = (_Float16*)alloc(WB2);
  _Float16* Wa1v = (_Float16*)alloc(WB2);
  _Float16* Wa1o = (_Float16*)alloc(WB2);
  _Float16* Wa2q = (_Float16*)alloc(WB2);
  _Float16* Wa2k = (_Float16*)alloc((size_t)KVD * DIMC * 2);
  _Float16* Wa2v = (_Float16*)alloc((size_t)KVD * DIMC * 2);
  _Float16* Wa2o = (_Float16*)alloc(WB2);
  _Float16* Wfc0 = (_Float16*)alloc((size_t)DIMC * 2560 * 2);
  _Float16* Wfc1 = (_Float16*)alloc((size_t)1280 * DIMC * 2);
  _Float16* Wcout = (_Float16*)alloc(WB2);

  float* mu   = (float*)alloc(64 * 4);
  float* rsig = (float*)alloc(64 * 4);

  const size_t ACT16 = (size_t)NB * SEQL * DIMC * 2;
  _Float16* act_a = (_Float16*)alloc(ACT16);   // xn / ln / final f16 x
  _Float16* act_b = (_Float16*)alloc(ACT16);   // attn outputs
  float* xbuf = (float*)alloc((size_t)NB * SEQL * DIMC * 4);

  const size_t QKV = (size_t)NB * HEADS * SEQL * DHP * 2;
  _Float16* qp = (_Float16*)alloc(QKV);
  _Float16* kp = (_Float16*)alloc(QKV);
  _Float16* vTb = (_Float16*)alloc(QKV);
  _Float16* kv16 = (_Float16*)alloc((size_t)NB * KVROWS * KVD * 2);
  const size_t KV2 = (size_t)NB * HEADS * LKP2 * DHP * 2;
  _Float16* kp2 = (_Float16*)alloc(KV2);
  _Float16* vT2 = (_Float16*)alloc(KV2);

  _Float16* g16 = (_Float16*)alloc((size_t)NB * SEQL * 2560 * 2);
  _Float16* h16 = (_Float16*)alloc((size_t)NB * SEQL * 1280 * 2);

  // zero the padded q/k/v buffers each launch (ws may be re-poisoned)
  hipMemsetAsync(qp, 0, QKV, stream);
  hipMemsetAsync(kp, 0, QKV, stream);
  hipMemsetAsync(vTb, 0, QKV, stream);
  hipMemsetAsync(kp2, 0, KV2, stream);
  hipMemsetAsync(vT2, 0, KV2, stream);
  // denominator ones-column in transposed V
  vt_ones_kernel<<<(NB * HEADS * SEQL + 255) / 256, 256, 0, stream>>>(vTb, SEQL);
  vt_ones_kernel<<<(NB * HEADS * LKP2 + 255) / 256, 256, 0, stream>>>(vT2, LKP2);

  auto packW = [&](const float* w, _Float16* dst, int K, int N) {
    int total = K * N;
    pack_b_kernel<<<(total + 255) / 256, 256, 0, stream>>>(w, dst, K, N);
  };
  packW(cin_w, Wcin, DIMC, DIMC);
  packW(a1_wq, Wa1q, DIMC, DIMC);
  packW(a1_wk, Wa1k, DIMC, DIMC);
  packW(a1_wv, Wa1v, DIMC, DIMC);
  packW(a1_wo, Wa1o, DIMC, DIMC);
  packW(a2_wq, Wa2q, DIMC, DIMC);
  packW(a2_wk, Wa2k, KVD, DIMC);
  packW(a2_wv, Wa2v, KVD, DIMC);
  packW(a2_wo, Wa2o, DIMC, DIMC);
  packW(fc0_w, Wfc0, DIMC, 2560);
  packW(fc1_w, Wfc1, 1280, DIMC);
  packW(cout_w, Wcout, DIMC, DIMC);

  // GroupNorm + transpose to f16 (b,hw,c)
  gn_stats_kernel<<<NB * 32, 256, 0, stream>>>(q_in, mu, rsig, 10 * SEQL, 1e-6f);
  gn_apply_kernel<<<dim3((DIMC * SEQL) / 256, 1, NB), 256, 0, stream>>>(
      q_in, mu, rsig, gn_s, gn_b, act_a);

  const long long sAct = (long long)SEQL * DIMC;
  const dim3 g320(DIMC / 64, SEQL / 256, NB);
  const float scale = 0.1581138830084190f;  // 1/sqrt(40)

  // conv_in (1x1) -> xbuf f32
  gemm16_kernel<EPI_F32><<<g320, 256, 0, stream>>>(
      act_a, Wcin, cin_b, nullptr, xbuf, SEQL, DIMC, DIMC, SEQL, sAct, sAct, 0, 0);

  // ---- self attention ----
  layernorm_kernel<<<NB * SEQL, 128, 0, stream>>>(xbuf, ln0_s, ln0_b, act_a, 1e-5f);
  gemm16_kernel<EPI_QK><<<g320, 256, 0, stream>>>(
      act_a, Wa1q, nullptr, nullptr, qp, SEQL, DIMC, DIMC, SEQL, sAct, 0, 0, SEQL);
  gemm16_kernel<EPI_QK><<<g320, 256, 0, stream>>>(
      act_a, Wa1k, nullptr, nullptr, kp, SEQL, DIMC, DIMC, SEQL, sAct, 0, 0, SEQL);
  gemm16_kernel<EPI_VT><<<g320, 256, 0, stream>>>(
      act_a, Wa1v, nullptr, nullptr, vTb, SEQL, DIMC, DIMC, SEQL, sAct, 0, 0, SEQL);
  attn_kernel<false><<<dim3(SEQL / 64, HEADS, NB), 128, 0, stream>>>(
      qp, kp, vTb, act_b, SEQL, SEQL, scale);
  gemm16_kernel<EPI_F32><<<g320, 256, 0, stream>>>(
      act_b, Wa1o, a1_bo, xbuf, xbuf, SEQL, DIMC, DIMC, SEQL, sAct, sAct, sAct, 0);

  // ---- cross attention ----
  layernorm_kernel<<<NB * SEQL, 128, 0, stream>>>(xbuf, ln1_s, ln1_b, act_a, 1e-5f);
  kv_cvt_kernel<<<(NB * KVROWS * KVD + 255) / 256, 256, 0, stream>>>(kv, kv16);
  gemm16_kernel<EPI_QK><<<g320, 256, 0, stream>>>(
      act_a, Wa2q, nullptr, nullptr, qp, SEQL, DIMC, DIMC, SEQL, sAct, 0, 0, SEQL);
  const long long sKV = (long long)KVROWS * KVD;
  gemm16_kernel<EPI_QK><<<dim3(DIMC / 64, 1, NB), 256, 0, stream>>>(
      kv16, Wa2k, nullptr, nullptr, kp2, KVROWS, DIMC, KVD, KVROWS, sKV, 0, 0, LKP2);
  gemm16_kernel<EPI_VT><<<dim3(DIMC / 64, 1, NB), 256, 0, stream>>>(
      kv16, Wa2v, nullptr, nullptr, vT2, KVROWS, DIMC, KVD, KVROWS, sKV, 0, 0, LKP2);
  attn_kernel<true><<<dim3(SEQL / 64, HEADS, NB), 128, 0, stream>>>(
      qp, kp2, vT2, act_b, 77, LKP2, scale);
  gemm16_kernel<EPI_F32><<<g320, 256, 0, stream>>>(
      act_b, Wa2o, a2_bo, xbuf, xbuf, SEQL, DIMC, DIMC, SEQL, sAct, sAct, sAct, 0);

  // ---- GEGLU MLP ----
  layernorm_kernel<<<NB * SEQL, 128, 0, stream>>>(xbuf, lnA_s, lnA_b, act_a, 1e-5f);
  gemm16_kernel<EPI_F16><<<dim3(2560 / 64, SEQL / 256, NB), 256, 0, stream>>>(
      act_a, Wfc0, fc0_b, nullptr, g16, SEQL, 2560, DIMC, SEQL,
      sAct, (long long)SEQL * 2560, 0, 0);
  geglu_kernel<<<dim3((SEQL * 1280) / 256, 1, NB), 256, 0, stream>>>(g16, h16);
  gemm16_kernel<EPI_F16><<<g320, 256, 0, stream>>>(
      h16, Wfc1, fc1_b, xbuf, act_a, SEQL, DIMC, 1280, SEQL,
      (long long)SEQL * 1280, sAct, sAct, 0);

  // conv_out (1x1) transposed + bias + res1(q) -> d_out (b,c,hw) f32
  gemm16_kernel<EPI_TOUT><<<g320, 256, 0, stream>>>(
      act_a, Wcout, cout_b, q_in, d_out, SEQL, DIMC, DIMC, SEQL, sAct, 0, 0, SEQL);
}
